// GraphormerAttentionHead_24747601559629
// MI455X (gfx1250) — matrix operand order repaired and checked by reading, not compiled, not run on hardware
//
#include <hip/hip_runtime.h>

#define NTOK 8192
#define DMODEL 512

typedef __attribute__((ext_vector_type(16))) _Float16 v16h;
typedef __attribute__((ext_vector_type(8)))  _Float16 v8h;
typedef __attribute__((ext_vector_type(4)))  _Float16 v4h;
typedef __attribute__((ext_vector_type(8)))  float    v8f;
typedef __attribute__((ext_vector_type(4)))  float    v4f;

// ---------------------------------------------------------------------------
// Fragment helpers (wave32, v_wmma_f32_16x16x32_f16 layouts per CDNA5 ISA 7.12.2)
// ---------------------------------------------------------------------------
__device__ __forceinline__ v16h cat16(v8h lo, v8h hi) {
  v16h r;
#pragma unroll
  for (int i = 0; i < 8; ++i) { r[i] = lo[i]; r[i + 8] = hi[i]; }
  return r;
}

// A-matrix fragment 16x32 (MxK), src row-major with given stride (halves):
// lanes 0-15: row=lane, K = {0..7, 16..23}; lanes 16-31: row=lane-16, K = {8..15, 24..31}
__device__ __forceinline__ v16h load_frag_a(const _Float16* base, int stride, int lane) {
  const int r = lane & 15;
  const int o = (lane < 16) ? 0 : 8;
  const _Float16* p = base + r * stride + o;
  v8h lo = *(const v8h*)(p);
  v8h hi = *(const v8h*)(p + 16);
  return cat16(lo, hi);
}

// B-matrix fragment 32x16 (KxN) given B^T row-major (rows = output columns,
// reduction dim contiguous): lanes 0-15: col=lane, K = {0..7, 16..23}; lanes 16-31: col=lane-16, K = {8..15, 24..31}
__device__ __forceinline__ v16h load_frag_bt(const _Float16* base, int stride, int lane) {
  const int r = lane & 15;
  const int o = (lane < 16) ? 0 : 8;
  const _Float16* p = base + r * stride + o;
  v8h lo = *(const v8h*)(p);
  v8h hi = *(const v8h*)(p + 16);
  return cat16(lo, hi);
}

__device__ __forceinline__ v8f wmma_f32_f16(v16h a, v16h b, v8f c) {
  return __builtin_amdgcn_wmma_f32_16x16x32_f16(false, a, false, b, (short)0, c,
                                                false, false);
}

// Async copy of one 32KiB K chunk (32 keys x 512 d, f16) global -> LDS.
// 128 threads x 256B each = 16 x b128 async loads per thread (ASYNCcnt).
// LDS byte address = low 32 bits of the generic pointer (LDS aperture).
__device__ __forceinline__ void async_copy_k_chunk(const _Float16* gsrc,
                                                   _Float16* lds_dst, int tid) {
  unsigned long long g =
      (unsigned long long)(const void*)gsrc + (unsigned long long)tid * 256ull;
  unsigned l = (unsigned)(unsigned long long)(void*)lds_dst + (unsigned)tid * 256u;
#pragma unroll
  for (int u = 0; u < 16; ++u) {
    asm volatile("global_load_async_to_lds_b128 %0, %1, off"
                 :: "v"(l + (unsigned)(u * 16)),
                    "v"(g + (unsigned long long)(u * 16))
                 : "memory");
  }
}

// ---------------------------------------------------------------------------
// fp32 -> fp16 cast (vectorized x4)
// ---------------------------------------------------------------------------
__global__ void cast_f32_to_f16(const float* __restrict__ src,
                                _Float16* __restrict__ dst, int n4) {
  int i = blockIdx.x * blockDim.x + threadIdx.x;
  if (i >= n4) return;
  float4 v = ((const float4*)src)[i];
  v4h h;
  h[0] = (_Float16)v.x; h[1] = (_Float16)v.y;
  h[2] = (_Float16)v.z; h[3] = (_Float16)v.w;
  ((v4h*)dst)[i] = h;
}

// ---------------------------------------------------------------------------
// Fused Q/K/V projection, software-pipelined: fragments for chunk c+1 are
// issued before the WMMAs of chunk c so loads overlap the matrix pipe.
// Outputs: qh, kh row-major f16; v stored transposed (vT[d][n]).
// ---------------------------------------------------------------------------
__global__ void __launch_bounds__(128) qkv_gemm(
    const _Float16* __restrict__ xh,
    const _Float16* __restrict__ wq, const _Float16* __restrict__ wk,
    const _Float16* __restrict__ wv,
    const float* __restrict__ bq, const float* __restrict__ bk,
    const float* __restrict__ bv,
    _Float16* __restrict__ qh, _Float16* __restrict__ kh,
    _Float16* __restrict__ vT) {
  const int lane = threadIdx.x & 31;
  const int wave = threadIdx.x >> 5;
  const int tile = blockIdx.x * 4 + wave;
  const int i0 = (tile >> 5) << 4;  // token tile
  const int j0 = (tile & 31) << 4;  // feature tile

  v8f cq = {}, ck = {}, cv = {};
  const _Float16* xa  = xh + (long)i0 * DMODEL;
  const _Float16* wqb = wq + (long)j0 * DMODEL;
  const _Float16* wkb = wk + (long)j0 * DMODEL;
  const _Float16* wvb = wv + (long)j0 * DMODEL;

  v16h a  = load_frag_a(xa, DMODEL, lane);
  v16h fq = load_frag_bt(wqb, DMODEL, lane);
  v16h fk = load_frag_bt(wkb, DMODEL, lane);
  v16h fv = load_frag_bt(wvb, DMODEL, lane);
#pragma unroll
  for (int c = 0; c < 16; ++c) {
    const int kn = (c + 1 < 16) ? (c + 1) * 32 : 0;  // last prefetch is dummy
    v16h an  = load_frag_a(xa + kn, DMODEL, lane);
    v16h fqn = load_frag_bt(wqb + kn, DMODEL, lane);
    v16h fkn = load_frag_bt(wkb + kn, DMODEL, lane);
    v16h fvn = load_frag_bt(wvb + kn, DMODEL, lane);
    cq = wmma_f32_f16(a, fq, cq);
    ck = wmma_f32_f16(a, fk, ck);
    cv = wmma_f32_f16(a, fv, cv);
    a = an; fq = fqn; fk = fkn; fv = fvn;
  }

  const int colLane = lane & 15;
  const int rowOff = (lane < 16) ? 0 : 8;
  const int j = j0 + colLane;
  const float vbq = bq[j], vbk = bk[j], vbv = bv[j];
#pragma unroll
  for (int r = 0; r < 8; ++r) {
    const int i = i0 + r + rowOff;
    qh[(long)i * DMODEL + j] = (_Float16)(cq[r] + vbq);
    kh[(long)i * DMODEL + j] = (_Float16)(ck[r] + vbk);
    vT[(long)j * NTOK + i]   = (_Float16)(cv[r] + vbv);
  }
}

// ---------------------------------------------------------------------------
// Flash attention. 1 block = 4 waves = one 16-query tile. Wave w owns output
// d-slice [w*128, w*128+128). The 32-key x 512-d K chunk is staged in LDS by
// CDNA5 async-copy (GLOBAL_LOAD_ASYNC_TO_LDS_B128 / s_wait_asynccnt), shared
// by all 4 waves. Per chunk:
//   P1: partial S over this wave's d-slice from LDS K (8 WMMA, software-
//       pipelined ds fragment loads), ds_add_f32 reduction; barrier.
//   P2: issue async prefetch of the next K chunk (overlaps encoding loads),
//       read full S + scale + spatial + edge, wait async, barrier.
//   softmax: online max/sum via 16-lane shfl; t=0 V-fragment load hoisted
//       above the exp/shfl section to hide global latency; P -> LDS.
//   P3: rescale O, accumulate P@V (8 WMMA, pipelined V-fragment loads).
// NxN matrices never touch HBM; encodings are read exactly once per WG.
// ---------------------------------------------------------------------------
__global__ void __launch_bounds__(128) attn_fused(
    const _Float16* __restrict__ qh, const _Float16* __restrict__ kh,
    const _Float16* __restrict__ vT,
    const float* __restrict__ sp, const float* __restrict__ ed,
    float* __restrict__ outp) {
  __shared__ __align__(16) _Float16 Qs[16 * DMODEL];   // 16 KiB
  __shared__ __align__(16) _Float16 Ks[32 * DMODEL];   // 32 KiB (async-filled)
  __shared__ __align__(16) float    Sacc[2][16 * 32];  // 4 KiB (double buffered)
  __shared__ __align__(16) _Float16 Pbuf[4][16 * 32];  // 4 KiB (wave-private)

  const int tid = threadIdx.x;
  const int lane = tid & 31;
  const int wave = tid >> 5;
  const int q0 = blockIdx.x * 16;
  const int d0 = wave * 128;
  const int colLane = lane & 15;
  const int rowOff = (lane < 16) ? 0 : 8;
  const float scale = 0.0441941738241592f;  // 1/sqrt(512)

  // Kick off async staging of K chunk 0 while we stage Q and zero S buffers.
  async_copy_k_chunk(kh, Ks, tid);
  {
    const v8h* src = (const v8h*)(qh + (long)q0 * DMODEL);
    v8h* dstv = (v8h*)Qs;
#pragma unroll
    for (int k = 0; k < 8; ++k) dstv[tid + k * 128] = src[tid + k * 128];
    float* z = &Sacc[0][0];
#pragma unroll
    for (int k = 0; k < 8; ++k) z[tid + k * 128] = 0.f;
  }
  asm volatile("s_wait_asynccnt 0" ::: "memory");
  __syncthreads();

  v8f o[8] = {};
  float m_r[8], l_r[8];
#pragma unroll
  for (int r = 0; r < 8; ++r) { m_r[r] = -1e30f; l_r[r] = 0.f; }

  _Float16* pb = Pbuf[wave];

  for (int kt = 0; kt < NTOK / 32; ++kt) {
    const int kc = kt * 32;
    float* sb = &Sacc[kt & 1][0];

    // ---- Phase 1: partial S over this wave's d-slice (K from LDS) ----
    v8f s0 = {}, s1 = {};
    v16h a  = load_frag_a(Qs + d0, DMODEL, lane);
    v16h b0 = load_frag_bt(Ks + d0, DMODEL, lane);
    v16h b1 = load_frag_bt(Ks + 16 * DMODEL + d0, DMODEL, lane);
#pragma unroll
    for (int c = 0; c < 4; ++c) {
      const int dkn = d0 + ((c + 1) & 3) * 32;  // c==3 reloads c=0 (dummy)
      v16h an  = load_frag_a(Qs + dkn, DMODEL, lane);
      v16h b0n = load_frag_bt(Ks + dkn, DMODEL, lane);
      v16h b1n = load_frag_bt(Ks + 16 * DMODEL + dkn, DMODEL, lane);
      s0 = wmma_f32_f16(a, b0, s0);
      s1 = wmma_f32_f16(a, b1, s1);
      a = an; b0 = b0n; b1 = b1n;
    }
#pragma unroll
    for (int r = 0; r < 8; ++r) {
      atomicAdd(&sb[(r + rowOff) * 32 + colLane], s0[r]);
      atomicAdd(&sb[(r + rowOff) * 32 + 16 + colLane], s1[r]);
    }
    __syncthreads();  // barrier1: all waves done reading Ks and adding S

    // ---- Phase 2: prefetch next K chunk; full S + encodings ----
    const int ktn = (kt + 1 < NTOK / 32) ? kt + 1 : kt;  // clamped (last is dummy)
    async_copy_k_chunk(kh + (long)(ktn * 32) * DMODEL, Ks, tid);

    float sv0[8], sv1[8];
#pragma unroll
    for (int r = 0; r < 8; ++r) {
      const int row = r + rowOff;
      const long eb = (long)(q0 + row) * NTOK + kc;
      const float e0 = sp[eb + colLane] + ed[eb + colLane];
      const float e1 = sp[eb + 16 + colLane] + ed[eb + 16 + colLane];
      sv0[r] = sb[row * 32 + colLane] * scale + e0;
      sv1[r] = sb[row * 32 + 16 + colLane] * scale + e1;
      __builtin_prefetch(sp + eb + 32, 0, 1);
      __builtin_prefetch(ed + eb + 32, 0, 1);
    }
    asm volatile("s_wait_asynccnt 0" ::: "memory");
    __syncthreads();  // barrier2: S reads done AND next K chunk resident
    // Recycle this S buffer (next use is kt+2, separated by kt+1's barriers).
    ((v4f*)sb)[tid] = v4f{};

    // Hoist the t=0 V fragment load above the softmax VALU section so the
    // global latency is hidden under exp/shfl work.
    v16h bv = load_frag_bt(vT + (long)d0 * NTOK + kc, NTOK, lane);

    // ---- Online softmax ----
    float corr[8];
#pragma unroll
    for (int r = 0; r < 8; ++r) {
      float v = fmaxf(sv0[r], sv1[r]);
#pragma unroll
      for (int off = 8; off >= 1; off >>= 1) v = fmaxf(v, __shfl_xor(v, off, 16));
      const float mn = fmaxf(m_r[r], v);
      corr[r] = __expf(m_r[r] - mn);
      const float p0 = __expf(sv0[r] - mn);
      const float p1 = __expf(sv1[r] - mn);
      m_r[r] = mn;
      const int row = r + rowOff;
      pb[row * 32 + colLane] = (_Float16)p0;  // P staged in A-layout
      pb[row * 32 + 16 + colLane] = (_Float16)p1;
      float s = p0 + p1;
#pragma unroll
      for (int off = 8; off >= 1; off >>= 1) s += __shfl_xor(s, off, 16);
      l_r[r] = l_r[r] * corr[r] + s;
    }
    // Wave-local LDS RAW on Pbuf (same-wave DS ops are in-order; fence compiler)
    asm volatile("s_wait_dscnt 0" ::: "memory");

    // ---- Phase 3: rescale O, accumulate P@V (pipelined V fragments) ----
    v16h pa = load_frag_a(pb, 32, lane);  // P 16x32 A-fragment
#pragma unroll
    for (int t = 0; t < 8; ++t) {
      v16h bvn = bv;
      if (t < 7)
        bvn = load_frag_bt(vT + (long)(d0 + (t + 1) * 16) * NTOK + kc, NTOK, lane);
      v8f c = o[t];
#pragma unroll
      for (int r = 0; r < 8; ++r) c[r] *= corr[r];
      o[t] = wmma_f32_f16(pa, bv, c);
      bv = bvn;
    }
  }

  // ---- Epilogue: O / l ----
#pragma unroll
  for (int r = 0; r < 8; ++r) {
    const float inv = 1.0f / l_r[r];
    const int row = q0 + r + rowOff;
#pragma unroll
    for (int t = 0; t < 8; ++t)
      outp[(long)row * DMODEL + d0 + t * 16 + colLane] = o[t][r] * inv;
  }
}

// ---------------------------------------------------------------------------
extern "C" void kernel_launch(void* const* d_in, const int* in_sizes, int n_in,
                              void* d_out, int out_size, void* d_ws,
                              size_t ws_size, hipStream_t stream) {
  const float* x  = (const float*)d_in[0];
  const float* sp = (const float*)d_in[1];
  const float* ed = (const float*)d_in[2];
  const float* Wq = (const float*)d_in[3];
  const float* bq = (const float*)d_in[4];
  const float* Wk = (const float*)d_in[5];
  const float* bk = (const float*)d_in[6];
  const float* Wv = (const float*)d_in[7];
  const float* bv = (const float*)d_in[8];
  float* outp = (float*)d_out;

  char* ws = (char*)d_ws;
  _Float16* xh  = (_Float16*)(ws);                         // 8 MiB
  _Float16* qh  = (_Float16*)(ws + (8u  << 20));           // 8 MiB
  _Float16* kh  = (_Float16*)(ws + (16u << 20));           // 8 MiB
  _Float16* vT  = (_Float16*)(ws + (24u << 20));           // 8 MiB (transposed V)
  _Float16* wqh = (_Float16*)(ws + (32u << 20));           // 512 KiB
  _Float16* wkh = (_Float16*)(ws + (32u << 20) + (512u << 10));
  _Float16* wvh = (_Float16*)(ws + (32u << 20) + (1024u << 10));

  const int nx4 = NTOK * DMODEL / 4;
  const int nw4 = DMODEL * DMODEL / 4;
  cast_f32_to_f16<<<(nx4 + 255) / 256, 256, 0, stream>>>(x, xh, nx4);
  cast_f32_to_f16<<<(nw4 + 255) / 256, 256, 0, stream>>>(Wq, wqh, nw4);
  cast_f32_to_f16<<<(nw4 + 255) / 256, 256, 0, stream>>>(Wk, wkh, nw4);
  cast_f32_to_f16<<<(nw4 + 255) / 256, 256, 0, stream>>>(Wv, wvh, nw4);

  // (N/16)*(D/16) = 16384 output tiles, 4 waves per block
  qkv_gemm<<<(NTOK / 16) * (DMODEL / 16) / 4, 128, 0, stream>>>(
      xh, wqh, wkh, wvh, bq, bk, bv, qh, kh, vT);

  attn_fused<<<NTOK / 16, 128, 0, stream>>>(qh, kh, vT, sp, ed, outp);
}
